// GCNNet_72679436582984
// MI455X (gfx1250) — matrix operand order repaired and checked by reading, not compiled, not run on hardware
//
#include <hip/hip_runtime.h>
#include <hip/hip_bf16.h>

typedef __attribute__((ext_vector_type(16))) _Float16 v16h;
typedef __attribute__((ext_vector_type(8)))  _Float16 v8h;
typedef __attribute__((ext_vector_type(8)))  float    v8f;

#define NNODES 250000
#define NEDGES 1000000
#define NBATCH 500
#define NBPAD  512     // NBATCH rounded up to M-tile grid

// ---------------------------------------------------------------- helpers
static __device__ inline v16h cat16(v8h a, v8h b) {
  return __builtin_shufflevector(a, b, 0,1,2,3,4,5,6,7,8,9,10,11,12,13,14,15);
}
static __device__ inline v8f zero8f() {
  v8f z = {0.f,0.f,0.f,0.f,0.f,0.f,0.f,0.f};
  return z;
}
// A fragment for v_wmma_f32_16x16x32_f16:
// lanes0-15 hold K=k..k+15, lanes16-31 hold K=k+16..k+31
static __device__ inline v16h loadA(const _Float16* ar, int kb, bool hi) {
  int k0 = kb + (hi ? 16 : 0);
  return cat16(*(const v8h*)(ar + k0), *(const v8h*)(ar + k0 + 8));
}
// B fragment: lane's column, 16 consecutive K values (lanes16-31 offset by 16)
static __device__ inline v16h loadB(const _Float16* bc, int kb, bool hi) {
  int kbB = kb + (hi ? 16 : 0);
  return cat16(*(const v8h*)(bc + kbB), *(const v8h*)(bc + kbB + 8));
}

// ---------------------------------------------------------------- small kernels
__global__ void k_fill_f32(float* p, float v, long long n) {
  long long i = blockIdx.x * (long long)blockDim.x + threadIdx.x;
  if (i < n) p[i] = v;
}
__global__ void k_degree(const int* dst, float* deg, int e) {
  int i = blockIdx.x * blockDim.x + threadIdx.x;
  if (i < e) atomicAdd(&deg[dst[i]], 1.0f);
}
__global__ void k_dinv(float* deg, int n) {
  int i = blockIdx.x * blockDim.x + threadIdx.x;
  if (i < n) deg[i] = rsqrtf(deg[i] + 1.0f);
}
__global__ void k_norm(const int* src, const int* dst, const float* dinv, float* nrm, int e) {
  int i = blockIdx.x * blockDim.x + threadIdx.x;
  if (i < e) nrm[i] = dinv[src[i]] * dinv[dst[i]];
}
// f32 [R,C] -> f16 [R,Cpad] (zero padded columns)
__global__ void k_pack(const float* s, _Float16* d, int R, int C, int Cpad) {
  long long i = blockIdx.x * (long long)blockDim.x + threadIdx.x;
  if (i >= (long long)R * Cpad) return;
  int r = (int)(i / Cpad), c = (int)(i % Cpad);
  d[i] = (c < C) ? (_Float16)s[(size_t)r * C + c] : (_Float16)0.f;
}
// f32 [K,N] -> f16 [N,Kpad] (transposed, zero padded K)
__global__ void k_packT(const float* s, _Float16* d, int K, int N, int Kpad) {
  long long i = blockIdx.x * (long long)blockDim.x + threadIdx.x;
  if (i >= (long long)N * Kpad) return;
  int n = (int)(i / Kpad), k = (int)(i % Kpad);
  d[i] = (k < K) ? (_Float16)s[(size_t)k * N + n] : (_Float16)0.f;
}

// ---------------------------------------------------------------- WMMA GEMMs
// One wave computes a 16x64 strip: 4 accumulators share one A fragment per K-step.
// A packed f16 [Mgrid, Kpad] (all tile rows readable); Bt packed f16 [Ngrid64, Kpad].
// Unpredicated loads; bounds applied only at the stores.
#define GEMM_BODY(STORE)                                                        \
  int lane = threadIdx.x & 31;                                                  \
  int gw = blockIdx.x * (blockDim.x >> 5) + (threadIdx.x >> 5);                 \
  int tilesM = (M + 15) >> 4;                                                   \
  int tilesN4 = (Nn + 63) >> 6;                                                 \
  if (gw >= tilesM * tilesN4) return;                                           \
  int tm = gw % tilesM, tn4 = gw / tilesM;                                      \
  bool hi = lane >= 16;                                                         \
  int row = tm * 16 + (lane & 15);                                              \
  int colb = tn4 * 64 + (lane & 15);                                            \
  const _Float16* ar  = A  + (size_t)row * Kpad;                                \
  const _Float16* bc0 = Bt + (size_t)(colb)      * Kpad;                        \
  const _Float16* bc1 = Bt + (size_t)(colb + 16) * Kpad;                        \
  const _Float16* bc2 = Bt + (size_t)(colb + 32) * Kpad;                        \
  const _Float16* bc3 = Bt + (size_t)(colb + 48) * Kpad;                        \
  v8f a0 = zero8f(), a1 = zero8f(), a2 = zero8f(), a3 = zero8f();               \
  for (int kb = 0; kb < Kpad; kb += 32) {                                       \
    __builtin_prefetch((const void*)(ar + kb + 512), 0, 0);                     \
    v16h av = loadA(ar, kb, hi);                                                \
    v16h b0 = loadB(bc0, kb, hi);                                               \
    v16h b1 = loadB(bc1, kb, hi);                                               \
    v16h b2 = loadB(bc2, kb, hi);                                               \
    v16h b3 = loadB(bc3, kb, hi);                                               \
    a0 = __builtin_amdgcn_wmma_f32_16x16x32_f16(false, av, false, b0, (short)0, a0, false, false); \
    a1 = __builtin_amdgcn_wmma_f32_16x16x32_f16(false, av, false, b1, (short)0, a1, false, false); \
    a2 = __builtin_amdgcn_wmma_f32_16x16x32_f16(false, av, false, b2, (short)0, a2, false, false); \
    a3 = __builtin_amdgcn_wmma_f32_16x16x32_f16(false, av, false, b3, (short)0, a3, false, false); \
  }                                                                             \
  _Pragma("unroll") for (int s = 0; s < 4; s++) {                               \
    v8f acc = (s == 0) ? a0 : (s == 1) ? a1 : (s == 2) ? a2 : a3;               \
    int col = tn4 * 64 + s * 16 + (lane & 15);                                  \
    _Pragma("unroll") for (int r = 0; r < 8; r++) {                             \
      int m = tm * 16 + r + (hi ? 8 : 0);                                       \
      if (m < M && col < Nn) { STORE; }                                         \
    }                                                                           \
  }

// C(f32)[M,N] = A @ B
__global__ void k_gemm_f32(const _Float16* __restrict__ A, const _Float16* __restrict__ Bt,
                           float* __restrict__ C, int M, int Nn, int Kpad) {
  GEMM_BODY(C[(size_t)m * Nn + col] = acc[r])
}
// C(f16)[M, ldc] at column offset, +bias, optional relu
__global__ void k_gemm_f16(const _Float16* __restrict__ A, const _Float16* __restrict__ Bt,
                           const float* __restrict__ bias, _Float16* __restrict__ C,
                           int M, int Nn, int Kpad, int ldc, int coloff, int relu) {
  GEMM_BODY({
    float v = acc[r] + bias[col];
    if (relu) v = fmaxf(v, 0.f);
    C[(size_t)m * ldc + coloff + col] = (_Float16)v;
  })
}

// Protein Conv1d as WMMA GEMM per batch element:
// c[b,o,p] = sum_{i,k} conv_w[o,i,k] * embp[target[b,i], p+k]
// A = conv_w f16 [32, 8000]; B gathered from padded f16 emb table [26,160] (no predication).
__global__ void k_conv_wmma(const _Float16* __restrict__ Aw, const int* __restrict__ target,
                            const _Float16* __restrict__ embp, const float* __restrict__ convb,
                            _Float16* __restrict__ cpack) {
  int lane = threadIdx.x & 31;
  int gw = blockIdx.x * (blockDim.x >> 5) + (threadIdx.x >> 5);
  if (gw >= NBATCH * 2 * 8) return;
  int b = gw >> 4;
  int tm = (gw >> 3) & 1;           // 2 tiles of output channels (32)
  int tn = gw & 7;                  // 8 tiles of positions (121 -> 128)
  int p = tn * 16 + (lane & 15);    // output position (gathers padded-safe up to 134<160)
  int orow = tm * 16 + (lane & 15);
  bool hi = lane >= 16;
  const _Float16* ar = Aw + (size_t)orow * 8000;
  const int* tg = target + (size_t)b * 1000;
  v8f acc = zero8f();
  for (int kb = 0; kb < 8000; kb += 32) {
    v16h av = loadA(ar, kb, hi);
    v16h bv;
    int kbB = kb + (hi ? 16 : 0);
#pragma unroll
    for (int e = 0; e < 16; e += 8) {       // 8 consecutive kk share one token pair
      int kk = kbB + e;
      int ii = kk >> 3;
      int tok = tg[ii];
      const _Float16* er = embp + (size_t)tok * 160 + p;
#pragma unroll
      for (int q = 0; q < 8; q++) bv[e + q] = er[(kk + q) & 7];
    }
    acc = __builtin_amdgcn_wmma_f32_16x16x32_f16(false, av, false, bv,
                                                 (short)0, acc, false, false);
  }
#pragma unroll
  for (int r = 0; r < 8; r++) {
    int o = tm * 16 + r + (hi ? 8 : 0);
    if (p < 121) {
      float v = acc[r] + convb[o];
      cpack[(size_t)b * 3872 + o * 121 + p] = (_Float16)v;
    }
  }
}

// ---------------------------------------------------------------- graph kernels
__global__ void k_edge_agg(const float* __restrict__ xw, const int* __restrict__ src,
                           const int* __restrict__ dst, const float* __restrict__ nrm,
                           float* __restrict__ agg, int M, long long total) {
  long long t = blockIdx.x * (long long)blockDim.x + threadIdx.x;
  if (t >= total) return;
  int e = (int)(t / M), f = (int)(t % M);
  atomicAdd(&agg[(size_t)dst[e] * M + f], xw[(size_t)src[e] * M + f] * nrm[e]);
}
__global__ void k_finish(const float* __restrict__ agg, const float* __restrict__ xw,
                         const float* __restrict__ dinv, const float* __restrict__ bias,
                         _Float16* __restrict__ outp, int M, int ld, long long total) {
  long long t = blockIdx.x * (long long)blockDim.x + threadIdx.x;
  if (t >= total) return;
  int i = (int)(t / M), f = (int)(t % M);
  float di = dinv[i];
  float v = agg[t] + xw[t] * di * di + bias[f];
  outp[(size_t)i * ld + f] = (_Float16)fmaxf(v, 0.f);
}
__global__ void k_finish_pool(const float* __restrict__ agg, const float* __restrict__ xw,
                              const float* __restrict__ dinv, const float* __restrict__ bias,
                              const int* __restrict__ batch, unsigned* __restrict__ pool,
                              int M, long long total) {
  long long t = blockIdx.x * (long long)blockDim.x + threadIdx.x;
  if (t >= total) return;
  int i = (int)(t / M), f = (int)(t % M);
  float di = dinv[i];
  float v = fmaxf(agg[t] + xw[t] * di * di + bias[f], 0.f);
  atomicMax(&pool[(size_t)batch[i] * M + f], __float_as_uint(v));
}
__global__ void k_out(const _Float16* __restrict__ u2, const float* __restrict__ ow,
                      const float* __restrict__ ob, float* __restrict__ out, int Bn) {
  int r = blockIdx.x * blockDim.x + threadIdx.x;
  if (r >= Bn) return;
  float s = ob[0];
  for (int k = 0; k < 512; k++) s += (float)u2[(size_t)r * 512 + k] * ow[k];
  out[r] = s;
}

// ---------------------------------------------------------------- launch
#define LGRID(n) dim3((unsigned)(((n) + 255) / 256)), dim3(256), 0, stream

extern "C" void kernel_launch(void* const* d_in, const int* in_sizes, int n_in,
                              void* d_out, int out_size, void* d_ws, size_t ws_size,
                              hipStream_t stream) {
  const float* x      = (const float*)d_in[0];
  const int*   ei     = (const int*)d_in[1];
  const int*   src    = ei;
  const int*   dst    = ei + NEDGES;
  const int*   batch  = (const int*)d_in[2];
  const int*   target = (const int*)d_in[3];
  const float* W1 = (const float*)d_in[4];  const float* b1 = (const float*)d_in[5];
  const float* W2 = (const float*)d_in[6];  const float* b2 = (const float*)d_in[7];
  const float* W3 = (const float*)d_in[8];  const float* b3 = (const float*)d_in[9];
  const float* fcg1_w = (const float*)d_in[10]; const float* fcg1_b = (const float*)d_in[11];
  const float* fcg2_w = (const float*)d_in[12]; const float* fcg2_b = (const float*)d_in[13];
  const float* emb    = (const float*)d_in[14];
  const float* conv_w = (const float*)d_in[15]; const float* conv_b = (const float*)d_in[16];
  const float* fcxt_w = (const float*)d_in[17]; const float* fcxt_b = (const float*)d_in[18];
  const float* fc1_w  = (const float*)d_in[19]; const float* fc1_b  = (const float*)d_in[20];
  const float* fc2_w  = (const float*)d_in[21]; const float* fc2_b  = (const float*)d_in[22];
  const float* out_w  = (const float*)d_in[23]; const float* out_b  = (const float*)d_in[24];
  float* out = (float*)d_out;

  char* w = (char*)d_ws;
  auto alloc = [&](size_t bytes) -> char* {
    char* p = w; w += (bytes + 255) & ~(size_t)255; return p;
  };
  const long long N = NNODES, E = NEDGES;
  // A-operand buffers: rows padded to the tile grid so fragment loads are unpredicated.
  float*     dinv   = (float*)alloc(N * 4);
  float*     nrm    = (float*)alloc(E * 4);
  _Float16*  hp0    = (_Float16*)alloc(N * 96 * 2);            // packed x   (78 ->96)
  _Float16*  hp1    = (_Float16*)alloc(N * 96 * 2);            // packed h1  (78 ->96)
  _Float16*  hp2    = (_Float16*)alloc(N * 160 * 2);           // packed h2  (156->160)
  float*     xw     = (float*)alloc(N * 312 * 4);
  float*     agg    = (float*)alloc(N * 312 * 4);
  unsigned*  pool   = (unsigned*)alloc((size_t)NBATCH * 312 * 4);
  _Float16*  gpack  = (_Float16*)alloc((size_t)NBPAD * 320 * 2);
  _Float16*  ghid   = (_Float16*)alloc((size_t)NBPAD * 1024 * 2);
  _Float16*  xcpack = (_Float16*)alloc((size_t)NBPAD * 256 * 2);
  _Float16*  u1     = (_Float16*)alloc((size_t)NBPAD * 1024 * 2);
  _Float16*  u2     = (_Float16*)alloc((size_t)NBPAD * 512 * 2);
  _Float16*  cpack  = (_Float16*)alloc((size_t)NBPAD * 3872 * 2);
  // Bt buffers: rows padded to multiple of 64 (4-tile strips read unpredicated).
  _Float16*  w1t    = (_Float16*)alloc((size_t)128  * 96 * 2);
  _Float16*  w2t    = (_Float16*)alloc((size_t)192  * 96 * 2);
  _Float16*  w3t    = (_Float16*)alloc((size_t)320  * 160 * 2);
  _Float16*  fg1t   = (_Float16*)alloc((size_t)1024 * 320 * 2);
  _Float16*  fg2t   = (_Float16*)alloc((size_t)128  * 1024 * 2);
  _Float16*  convA  = (_Float16*)alloc((size_t)32   * 8000 * 2);
  _Float16*  fxtt   = (_Float16*)alloc((size_t)128  * 3872 * 2);
  _Float16*  f1t    = (_Float16*)alloc((size_t)1024 * 256 * 2);
  _Float16*  f2t    = (_Float16*)alloc((size_t)512  * 1024 * 2);
  _Float16*  embp   = (_Float16*)alloc((size_t)26   * 160 * 2); // padded emb table

  auto gemm_blocks = [](int M, int Nn) -> unsigned {
    long long waves = (long long)((M + 15) / 16) * ((Nn + 63) / 64);
    return (unsigned)((waves + 7) / 8);
  };

  // ---- degrees / norms
  k_fill_f32<<<LGRID(N)>>>(dinv, 0.f, N);
  k_degree<<<LGRID(E)>>>(dst, dinv, (int)E);
  k_dinv<<<LGRID(N)>>>(dinv, (int)N);
  k_norm<<<LGRID(E)>>>(src, dst, dinv, nrm, (int)E);

  // ---- pack activations / weights to f16 (K padded to mult. of 32)
  k_pack <<<LGRID(N * 96)>>>(x, hp0, (int)N, 78, 96);
  k_fill_f32<<<LGRID(N * 48)>>>((float*)hp1, 0.f, N * 48);   // zero pads of hp1
  k_fill_f32<<<LGRID(N * 80)>>>((float*)hp2, 0.f, N * 80);   // zero pads of hp2
  k_packT<<<LGRID(78  * 96)>>> (W1, w1t, 78, 78, 96);
  k_packT<<<LGRID(156 * 96)>>> (W2, w2t, 78, 156, 96);
  k_packT<<<LGRID(312 * 160)>>>(W3, w3t, 156, 312, 160);
  k_packT<<<LGRID(1024 * 320)>>>(fcg1_w, fg1t, 312, 1024, 320);
  k_packT<<<LGRID(128 * 1024)>>>(fcg2_w, fg2t, 1024, 128, 1024);
  k_pack <<<LGRID(32 * 8000)>>>(conv_w, convA, 32, 8000, 8000);
  k_packT<<<LGRID(128 * 3872)>>>(fcxt_w, fxtt, 3872, 128, 3872);
  k_packT<<<LGRID(1024 * 256)>>>(fc1_w, f1t, 256, 1024, 256);
  k_packT<<<LGRID(512 * 1024)>>>(fc2_w, f2t, 1024, 512, 1024);
  k_pack <<<LGRID(26 * 160)>>>(emb, embp, 26, 128, 160);

  // ---- GCN layer 1: 250000x78 @ 78x78
  k_gemm_f32<<<gemm_blocks((int)N, 78), 256, 0, stream>>>(hp0, w1t, xw, (int)N, 78, 96);
  k_fill_f32<<<LGRID(N * 78)>>>(agg, 0.f, N * 78);
  k_edge_agg<<<LGRID(E * 78)>>>(xw, src, dst, nrm, agg, 78, E * 78);
  k_finish<<<LGRID(N * 78)>>>(agg, xw, dinv, b1, hp1, 78, 96, N * 78);

  // ---- GCN layer 2: 250000x78 @ 78x156
  k_gemm_f32<<<gemm_blocks((int)N, 156), 256, 0, stream>>>(hp1, w2t, xw, (int)N, 156, 96);
  k_fill_f32<<<LGRID(N * 156)>>>(agg, 0.f, N * 156);
  k_edge_agg<<<LGRID(E * 156)>>>(xw, src, dst, nrm, agg, 156, E * 156);
  k_finish<<<LGRID(N * 156)>>>(agg, xw, dinv, b2, hp2, 156, 160, N * 156);

  // ---- GCN layer 3: 250000x156 @ 156x312, fused relu + max-pool
  k_gemm_f32<<<gemm_blocks((int)N, 312), 256, 0, stream>>>(hp2, w3t, xw, (int)N, 312, 160);
  k_fill_f32<<<LGRID(N * 312)>>>(agg, 0.f, N * 312);
  k_edge_agg<<<LGRID(E * 312)>>>(xw, src, dst, nrm, agg, 312, E * 312);
  k_fill_f32<<<LGRID((long long)NBATCH * 312)>>>((float*)pool, 0.f, (long long)NBATCH * 312);
  k_finish_pool<<<LGRID(N * 312)>>>(agg, xw, dinv, b3, batch, pool, 312, N * 312);

  // ---- graph head: relu(g@fcg1) @ fcg2 -> xc[:, 0:128]
  k_pack<<<LGRID((long long)NBATCH * 320)>>>((const float*)pool, gpack, NBATCH, 312, 320);
  k_gemm_f16<<<gemm_blocks(NBATCH, 1024), 256, 0, stream>>>(gpack, fg1t, fcg1_b, ghid,
                                                            NBATCH, 1024, 320, 1024, 0, 1);
  k_gemm_f16<<<gemm_blocks(NBATCH, 128), 256, 0, stream>>>(ghid, fg2t, fcg2_b, xcpack,
                                                           NBATCH, 128, 1024, 256, 0, 0);

  // ---- protein branch: conv1d (WMMA, im2col gather) then fcxt -> xc[:, 128:256]
  k_conv_wmma<<<(NBATCH * 16 + 7) / 8, 256, 0, stream>>>(convA, target, embp, conv_b, cpack);
  k_gemm_f16<<<gemm_blocks(NBATCH, 128), 256, 0, stream>>>(cpack, fxtt, fcxt_b, xcpack,
                                                           NBATCH, 128, 3872, 256, 128, 0);

  // ---- fusion head
  k_gemm_f16<<<gemm_blocks(NBATCH, 1024), 256, 0, stream>>>(xcpack, f1t, fc1_b, u1,
                                                            NBATCH, 1024, 256, 1024, 0, 1);
  k_gemm_f16<<<gemm_blocks(NBATCH, 512), 256, 0, stream>>>(u1, f2t, fc2_b, u2,
                                                           NBATCH, 512, 1024, 512, 0, 1);
  k_out<<<LGRID(NBATCH)>>>(u2, out_w, out_b, out, NBATCH);
}
